// Attention_57827439673725
// MI455X (gfx1250) — matrix operand order repaired and checked by reading, not compiled, not run on hardware
//
#include <hip/hip_runtime.h>
#include <hip/hip_bf16.h>

// ---------------------------------------------------------------------------
// CDNA5 (gfx1250) attention block, f16-resident pipeline:
//   prep: x -> f16; qkv_w/proj_w -> f16 transposed [N][K]
//   qkv(f16) = x @ qkv_w + qkv_b        (WMMA GEMM)
//   q,k = rope(rms_norm(q|k))           (f16 in/out, f32 math)
//   vT  = per-(b,h) transpose of v      (one-time pass)
//   o(f16) = flash_attention(q, k, v)   (WMMA, online softmax)
//   out(f32) = o @ proj_w + proj_b      (WMMA GEMM)
// LDS tile fills: global_load_async_to_lds_b128, DOUBLE-BUFFERED so stage i+1
// copies overlap stage i WMMAs (one barrier per stage). Fragment gathers are
// contiguous 16B runs -> ds_load_b128.
// ---------------------------------------------------------------------------

typedef __attribute__((ext_vector_type(16))) _Float16 v16h;
typedef __attribute__((ext_vector_type(8)))  _Float16 v8h;
typedef __attribute__((ext_vector_type(4)))  _Float16 v4h;
typedef __attribute__((ext_vector_type(8)))  float    v8f;
typedef __attribute__((ext_vector_type(4)))  float    f4;

#define B_SZ   32
#define N_SZ   1024
#define C_SZ   1024
#define H_SZ   16
#define HD_SZ  64

#if __has_builtin(__builtin_amdgcn_global_load_async_to_lds_b128)
#define USE_ASYNC_CP 1
#else
#define USE_ASYNC_CP 0
#endif

// Builtin takes typed int4 pointers: (AS1 src, AS3 dst, imm offset, imm cpol).
typedef int i32x4 __attribute__((vector_size(16)));
typedef __attribute__((address_space(1))) i32x4 gas_i32x4;
typedef __attribute__((address_space(3))) i32x4 las_i32x4;

// Copy 16 bytes global -> LDS (async when available).
__device__ __forceinline__ void cp16B(_Float16* ldst, const _Float16* gsrc) {
#if USE_ASYNC_CP
  __builtin_amdgcn_global_load_async_to_lds_b128(
      (gas_i32x4*)(unsigned long long)gsrc,
      (las_i32x4*)(unsigned)(unsigned long long)ldst,
      0, 0);
#else
  *(v8h*)ldst = *(const v8h*)gsrc;
#endif
}

__device__ __forceinline__ void wait_async() {
#if USE_ASYNC_CP
#if __has_builtin(__builtin_amdgcn_s_wait_asynccnt)
  __builtin_amdgcn_s_wait_asynccnt(0);
#else
  asm volatile("s_wait_asynccnt 0x0" ::: "memory");
#endif
#endif
}

__device__ __forceinline__ v8f wmma_f16(v16h a, v16h b, v8f c) {
  return __builtin_amdgcn_wmma_f32_16x16x32_f16(
      false, a, false, b, (short)0, c, false, false);
}

__device__ __forceinline__ v16h frag_cat(v8h lo, v8h hi) {
  return __builtin_shufflevector(lo, hi, 0, 1, 2, 3, 4, 5, 6, 7,
                                 8, 9, 10, 11, 12, 13, 14, 15);
}

// A fragment (16x32 f16) from a row-major f16 tile (LDS or global),
// row stride ld halfs (ld % 8 == 0, 16B-aligned rows). Lane (g,lr): row lr,
// K = {8g..8g+7} U {16+8g..23+8g} -> two contiguous b128 loads.
__device__ __forceinline__ v16h load_a_frag(const _Float16* base, int ld) {
  const int lane = threadIdx.x & 31;
  const int g = lane >> 4, lr = lane & 15;
  const _Float16* row = base + (size_t)lr * ld + 8 * g;
  return frag_cat(*(const v8h*)row, *(const v8h*)(row + 16));
}

// B fragment (32x16 f16) from a TRANSPOSED tile Bt[n][k], row stride ld halfs
// (ld % 8 == 0). Lane (g,lr): col n=lr, K = {8g..8g+7} U {16+8g..23+8g} -> two b128 loads.
__device__ __forceinline__ v16h load_bt_frag(const _Float16* base, int ld) {
  const int lane = threadIdx.x & 31;
  const int g = lane >> 4, lr = lane & 15;
  const _Float16* col = base + lr * ld + 8 * g;
  return frag_cat(*(const v8h*)col, *(const v8h*)(col + 16));
}

// ---------------------------------------------------------------------------
// Prep kernel A: f32 -> f16, vectorized (float4 -> 4x f16).
// ---------------------------------------------------------------------------
__global__ __launch_bounds__(256) void cvt_f32_f16(
    const float* __restrict__ src, _Float16* __restrict__ dst, long long n4) {
  const long long i = (long long)blockIdx.x * 256 + threadIdx.x;
  if (i >= n4) return;
  const f4 v = ((const f4*)src)[i];
  v4h h;
#pragma unroll
  for (int j = 0; j < 4; ++j) h[j] = (_Float16)v[j];
  ((v4h*)dst)[i] = h;
}

// ---------------------------------------------------------------------------
// Prep kernel B: src[R][Cc] f32 -> dst[Cc][R] f16 (tiled transpose).
// ---------------------------------------------------------------------------
__global__ __launch_bounds__(256) void transpose_f32_to_f16(
    const float* __restrict__ src, _Float16* __restrict__ dst, int R, int Cc) {
  __shared__ float tl[32][33];
  const int c0 = blockIdx.x * 32, r0 = blockIdx.y * 32;
  const int tr = threadIdx.x >> 5;  // 0..7
  const int tc = threadIdx.x & 31;
#pragma unroll
  for (int i = 0; i < 4; ++i)
    tl[tr + 8 * i][tc] = src[(size_t)(r0 + tr + 8 * i) * Cc + c0 + tc];
  __syncthreads();
#pragma unroll
  for (int i = 0; i < 4; ++i)
    dst[(size_t)(c0 + tr + 8 * i) * R + r0 + tc] = (_Float16)tl[tc][tr + 8 * i];
}

// ---------------------------------------------------------------------------
// Prep kernel C: per-(b,h) transpose of V: qkv[bn][2C + h*64 + d] f16
//   -> vt[(b*H+h)*64 + d][n] f16.
// ---------------------------------------------------------------------------
__global__ __launch_bounds__(256) void transpose_v_f16(
    const _Float16* __restrict__ qkv, _Float16* __restrict__ vt) {
  __shared__ _Float16 tl[32][33];
  const int bh = blockIdx.z;
  const int n0 = blockIdx.x * 32, d0 = blockIdx.y * 32;
  const int b = bh >> 4, h = bh & 15;
  const int tr = threadIdx.x >> 5, tc = threadIdx.x & 31;
#pragma unroll
  for (int i = 0; i < 4; ++i)
    tl[tr + 8 * i][tc] =
        qkv[(size_t)(b * N_SZ + n0 + tr + 8 * i) * 3072 + 2 * C_SZ +
            h * HD_SZ + d0 + tc];
  __syncthreads();
#pragma unroll
  for (int i = 0; i < 4; ++i)
    vt[((size_t)bh * HD_SZ + d0 + tr + 8 * i) * N_SZ + n0 + tc] =
        tl[tc][tr + 8 * i];
}

// ---------------------------------------------------------------------------
// GEMM: C[M,N] = A[M,K](f16) @ Bt[N,K](f16,transposed) + bias[N](f32).
// 256 threads = 8 waves, 128x128 tile, K-step 64; wave owns 64x32 = 4x2
// fragments, 16 WMMAs per K-stage. Double-buffered async LDS fills.
// ---------------------------------------------------------------------------
template <typename OutT>
__global__ __launch_bounds__(256) void gemm_bias_wmma(
    const _Float16* __restrict__ A, const _Float16* __restrict__ Bt,
    const float* __restrict__ bias, OutT* __restrict__ C,
    int M, int N, int K) {
  __shared__ __attribute__((aligned(16))) _Float16 As[2][128][72];  // m x k64
  __shared__ __attribute__((aligned(16))) _Float16 Bs[2][128][72];  // n x k64
  const int t = threadIdx.x;
  const int lane = t & 31, wid = t >> 5;
  const int g = lane >> 4, lr = lane & 15;
  const int m0 = blockIdx.y * 128, n0 = blockIdx.x * 128;
  const int mw = (wid & 1) * 64, nw = (wid >> 1) * 32;

  // Per-thread copy-chunk coordinates and streaming source pointers
  const _Float16* asrc[4];
  const _Float16* bsrc[4];
#pragma unroll
  for (int i = 0; i < 4; ++i) {
    const int idx = t + i * 256;             // 0..1023 16B chunks per tile
    const int r = idx >> 3, c8 = (idx & 7) * 8;
    asrc[i] = A + (size_t)(m0 + r) * K + c8;
    bsrc[i] = Bt + (size_t)(n0 + r) * K + c8;
  }

  // Prologue: stage 0 copies into buffer 0
#pragma unroll
  for (int i = 0; i < 4; ++i) {
    const int idx = t + i * 256;
    const int r = idx >> 3, c8 = (idx & 7) * 8;
    cp16B(&As[0][r][c8], asrc[i]);
    cp16B(&Bs[0][r][c8], bsrc[i]);
  }

  v8f acc[4][2] = {};
  int stage = 0;

  for (int kk = 0; kk < K; kk += 64, stage ^= 1) {
    wait_async();       // this wave's copies for tile (kk) have landed
    __syncthreads();    // all waves landed; prior reads of other buffer done

    if (kk + 64 < K) {  // overlap next tile's copies with this tile's WMMAs
#pragma unroll
      for (int i = 0; i < 4; ++i) {
        const int idx = t + i * 256;
        const int r = idx >> 3, c8 = (idx & 7) * 8;
        cp16B(&As[stage ^ 1][r][c8], asrc[i] + 64);
        cp16B(&Bs[stage ^ 1][r][c8], bsrc[i] + 64);
        asrc[i] += 64;
        bsrc[i] += 64;
      }
    }

#pragma unroll
    for (int ks = 0; ks < 2; ++ks) {
      const int k0 = ks * 32;
      const v16h b0 = load_bt_frag(&Bs[stage][nw][k0], 72);
      const v16h b1 = load_bt_frag(&Bs[stage][nw + 16][k0], 72);
#pragma unroll
      for (int i = 0; i < 4; ++i) {
        const v16h a = load_a_frag(&As[stage][mw + i * 16][k0], 72);
        acc[i][0] = wmma_f16(a, b0, acc[i][0]);
        acc[i][1] = wmma_f16(a, b1, acc[i][1]);
      }
    }
  }

#pragma unroll
  for (int j = 0; j < 2; ++j) {
    const float bv = bias[n0 + nw + j * 16 + lr];
#pragma unroll
    for (int i = 0; i < 4; ++i) {
#pragma unroll
      for (int v = 0; v < 8; ++v) {
        const int row = m0 + mw + i * 16 + v + 8 * g;  // C/D: M = v + 8*(lane/16)
        C[(size_t)row * N + n0 + nw + j * 16 + lr] = (OutT)(acc[i][j][v] + bv);
      }
    }
  }
}

// ---------------------------------------------------------------------------
// Fused RMSNorm + 2D RoPE on q,k (f16 storage, f32 math). One wave32 per
// 64-elem head vector; lane holds d and d+32 so rotate_half partner is local.
// ---------------------------------------------------------------------------
__global__ __launch_bounds__(256) void rms_rope_f16(
    _Float16* __restrict__ qkv, const float* __restrict__ q_gamma,
    const float* __restrict__ k_gamma) {
  const int lane = threadIdx.x & 31;
  const int wid = threadIdx.x >> 5;
  const long long idx = (long long)blockIdx.x * 8 + wid;  // B*N*H*2 vectors
  const int qk = (int)(idx & 1);
  const int h  = (int)((idx >> 1) & 15);
  const long long bn = idx >> 5;

  _Float16* p = qkv + (size_t)bn * 3072 + (qk ? C_SZ : 0) + h * HD_SZ;
  const float* gamma = qk ? k_gamma : q_gamma;

  float x0 = (float)p[lane];
  float x1 = (float)p[lane + 32];

  float ss = x0 * x0 + x1 * x1;
#pragma unroll
  for (int m = 16; m >= 1; m >>= 1) ss += __shfl_xor(ss, m, 32);
  const float inv = rsqrtf(ss * (1.0f / 64.0f) + 1e-6f);
  x0 *= inv * gamma[lane];
  x1 *= inv * gamma[lane + 32];

  const int n = (int)(bn & (N_SZ - 1));
  const float rpos = (float)(n >> 5);
  const float cpos = (float)(n & 31);
  const float fr = __powf(10000.0f, -(float)(lane & 15) * (1.0f / 16.0f));
  float s0, c0, s1, c1;
  __sincosf(rpos * fr, &s0, &c0);
  __sincosf(cpos * fr, &s1, &c1);
  p[lane]      = (_Float16)(x0 * c0 - x1 * s0);  // rh[d]    = -x[d+32]
  p[lane + 32] = (_Float16)(x1 * c1 + x0 * s1);  // rh[d+32] =  x[d]
}

// ---------------------------------------------------------------------------
// Flash attention. WG = 128 threads = 4 waves = 64 queries of one (b,h);
// 64-key blocks, double-buffered async K/V staging. K tile row-major (k-dim =
// contiguous head dim -> direct B-frags for QK^T); V pre-transposed in global.
// Online softmax in f32, 16-lane butterflies. Output f16.
// ---------------------------------------------------------------------------
__global__ __launch_bounds__(128) void flash_attn_wmma(
    const _Float16* __restrict__ qkv, const _Float16* __restrict__ vt,
    _Float16* __restrict__ out) {
  __shared__ __attribute__((aligned(16))) _Float16 Ks[2][64][72];  // key x d
  __shared__ __attribute__((aligned(16))) _Float16 Vt[2][64][72];  // d x key
  __shared__ __attribute__((aligned(16))) _Float16 Ps[4][16][72];  // row x key

  const int t = threadIdx.x, lane = t & 31, wid = t >> 5;
  const int g = lane >> 4, lr = lane & 15;
  const int bh = blockIdx.y;
  const int b = bh >> 4, h = bh & 15;
  const int q0 = blockIdx.x * 64 + wid * 16;
  const size_t rowbase = (size_t)b * N_SZ;

  // Q fragments, softmax scale 1/sqrt(64) folded in (exact in f16)
  const _Float16* qp = qkv + (rowbase + q0) * 3072 + h * HD_SZ;
  v16h qa0 = load_a_frag(qp, 3072);       // d 0..31
  v16h qa1 = load_a_frag(qp + 32, 3072);  // d 32..63
#pragma unroll
  for (int i = 0; i < 16; ++i) {
    qa0[i] *= (_Float16)0.125f;
    qa1[i] *= (_Float16)0.125f;
  }

  // Per-thread copy-chunk coordinates and streaming source pointers
  const _Float16* ksrc[4];
  const _Float16* vsrc[4];
#pragma unroll
  for (int i = 0; i < 4; ++i) {
    const int idx = t + i * 128;           // 0..511 16B chunks per tile
    const int r = idx >> 3, c8 = (idx & 7) * 8;
    ksrc[i] = qkv + (rowbase + r) * 3072 + C_SZ + h * HD_SZ + c8;  // r = key
    vsrc[i] = vt + ((size_t)bh * HD_SZ + r) * N_SZ + c8;           // r = d
  }

  // Prologue: key block 0 into buffer 0
#pragma unroll
  for (int i = 0; i < 4; ++i) {
    const int idx = t + i * 128;
    const int r = idx >> 3, c8 = (idx & 7) * 8;
    cp16B(&Ks[0][r][c8], ksrc[i]);
    cp16B(&Vt[0][r][c8], vsrc[i]);
  }

  v8f o0 = {}, o1 = {}, o2 = {}, o3 = {};
  float rm[8], rs[8];
#pragma unroll
  for (int v = 0; v < 8; ++v) { rm[v] = -1e30f; rs[v] = 0.0f; }

  int stage = 0;
  for (int j0 = 0; j0 < N_SZ; j0 += 64, stage ^= 1) {
    wait_async();
    __syncthreads();

    if (j0 + 64 < N_SZ) {  // overlap next K/V block with this block's compute
#pragma unroll
      for (int i = 0; i < 4; ++i) {
        const int idx = t + i * 128;
        const int r = idx >> 3, c8 = (idx & 7) * 8;
        cp16B(&Ks[stage ^ 1][r][c8], ksrc[i] + (size_t)64 * 3072);
        cp16B(&Vt[stage ^ 1][r][c8], vsrc[i] + 64);
        ksrc[i] += (size_t)64 * 3072;
        vsrc[i] += 64;
      }
    }

    // S = Q @ K^T : four 16x16 key tiles, two d-slices each
    v8f s[4] = {v8f{}, v8f{}, v8f{}, v8f{}};
#pragma unroll
    for (int kt = 0; kt < 4; ++kt) {
      const v16h kb0 = load_bt_frag(&Ks[stage][kt * 16][0], 72);
      const v16h kb1 = load_bt_frag(&Ks[stage][kt * 16][32], 72);
      s[kt] = wmma_f16(qa0, kb0, s[kt]);
      s[kt] = wmma_f16(qa1, kb1, s[kt]);
    }

    // Online softmax: row M = v + 8g, cols striped over 16 lanes
#pragma unroll
    for (int v = 0; v < 8; ++v) {
      float bm = fmaxf(fmaxf(s[0][v], s[1][v]), fmaxf(s[2][v], s[3][v]));
#pragma unroll
      for (int m = 1; m <= 8; m <<= 1) bm = fmaxf(bm, __shfl_xor(bm, m, 32));
      const float mn = fmaxf(rm[v], bm);
      const float alpha = __expf(rm[v] - mn);
      float p[4], bs = 0.0f;
#pragma unroll
      for (int kt = 0; kt < 4; ++kt) { p[kt] = __expf(s[kt][v] - mn); bs += p[kt]; }
#pragma unroll
      for (int m = 1; m <= 8; m <<= 1) bs += __shfl_xor(bs, m, 32);
      rs[v] = rs[v] * alpha + bs;
      rm[v] = mn;
      o0[v] *= alpha; o1[v] *= alpha; o2[v] *= alpha; o3[v] *= alpha;
      const int row = v + 8 * g;
#pragma unroll
      for (int kt = 0; kt < 4; ++kt) Ps[wid][row][kt * 16 + lr] = (_Float16)p[kt];
    }

    // O += P @ V : two key k-steps x four d fragments
    const v16h pa0 = load_a_frag(&Ps[wid][0][0], 72);   // keys 0..31
    const v16h pa1 = load_a_frag(&Ps[wid][0][32], 72);  // keys 32..63
    o0 = wmma_f16(pa0, load_bt_frag(&Vt[stage][0][0], 72), o0);
    o1 = wmma_f16(pa0, load_bt_frag(&Vt[stage][16][0], 72), o1);
    o2 = wmma_f16(pa0, load_bt_frag(&Vt[stage][32][0], 72), o2);
    o3 = wmma_f16(pa0, load_bt_frag(&Vt[stage][48][0], 72), o3);
    o0 = wmma_f16(pa1, load_bt_frag(&Vt[stage][0][32], 72), o0);
    o1 = wmma_f16(pa1, load_bt_frag(&Vt[stage][16][32], 72), o1);
    o2 = wmma_f16(pa1, load_bt_frag(&Vt[stage][32][32], 72), o2);
    o3 = wmma_f16(pa1, load_bt_frag(&Vt[stage][48][32], 72), o3);
  }

  // Normalize, write attention output as f16 [B*N, C]
#pragma unroll
  for (int v = 0; v < 8; ++v) {
    const float inv = 1.0f / rs[v];
    const size_t row = rowbase + q0 + v + 8 * g;
    _Float16* op = out + row * C_SZ + h * HD_SZ;
    op[lr]      = (_Float16)(o0[v] * inv);
    op[16 + lr] = (_Float16)(o1[v] * inv);
    op[32 + lr] = (_Float16)(o2[v] * inv);
    op[48 + lr] = (_Float16)(o3[v] * inv);
  }
}

// ---------------------------------------------------------------------------
extern "C" void kernel_launch(void* const* d_in, const int* in_sizes, int n_in,
                              void* d_out, int out_size, void* d_ws,
                              size_t ws_size, hipStream_t stream) {
  (void)in_sizes; (void)n_in; (void)out_size; (void)ws_size;
  const float* x       = (const float*)d_in[0];
  const float* qkv_w   = (const float*)d_in[1];
  const float* qkv_b   = (const float*)d_in[2];
  const float* proj_w  = (const float*)d_in[3];
  const float* proj_b  = (const float*)d_in[4];
  const float* q_gamma = (const float*)d_in[5];
  const float* k_gamma = (const float*)d_in[6];
  float* out = (float*)d_out;

  const size_t BN = (size_t)B_SZ * N_SZ;  // 32768
  _Float16* p = (_Float16*)d_ws;
  _Float16* xh    = p;  p += BN * C_SZ;                 // [BN][1024]
  _Float16* wq_t  = p;  p += (size_t)3 * C_SZ * C_SZ;   // [3072][1024]
  _Float16* wp_t  = p;  p += (size_t)C_SZ * C_SZ;       // [1024][1024]
  _Float16* qkvh  = p;  p += BN * 3 * C_SZ;             // [BN][3072]
  _Float16* vth   = p;  p += (size_t)B_SZ * H_SZ * HD_SZ * N_SZ;  // [512*64][1024]
  _Float16* attnh = p;                                   // [BN][1024]

  // Prep: x -> f16; weights -> f16 transposed [N][K]
  cvt_f32_f16<<<(unsigned)(BN * C_SZ / 4 / 256), 256, 0, stream>>>(
      x, xh, (long long)(BN * C_SZ / 4));
  transpose_f32_to_f16<<<dim3(3 * C_SZ / 32, C_SZ / 32), 256, 0, stream>>>(
      qkv_w, wq_t, C_SZ, 3 * C_SZ);
  transpose_f32_to_f16<<<dim3(C_SZ / 32, C_SZ / 32), 256, 0, stream>>>(
      proj_w, wp_t, C_SZ, C_SZ);

  // 1) qkv = x @ qkv_w + qkv_b   (f16 out)
  gemm_bias_wmma<_Float16><<<dim3(3 * C_SZ / 128, BN / 128), 256, 0, stream>>>(
      xh, wq_t, qkv_b, qkvh, (int)BN, 3 * C_SZ, C_SZ);

  // 2) q,k <- rope(rms_norm(.)) in place
  rms_rope_f16<<<(unsigned)(BN * H_SZ * 2 / 8), 256, 0, stream>>>(
      qkvh, q_gamma, k_gamma);

  // 2b) v -> per-(b,h) transposed copy for P@V fragment loads
  transpose_v_f16<<<dim3(N_SZ / 32, HD_SZ / 32, B_SZ * H_SZ), 256, 0, stream>>>(
      qkvh, vth);

  // 3) flash attention per (b,h)  (f16 out)
  flash_attn_wmma<<<dim3(N_SZ / 64, B_SZ * H_SZ), 128, 0, stream>>>(
      qkvh, vth, attnh);

  // 4) out = attn @ proj_w + proj_b  (f32 out)
  gemm_bias_wmma<float><<<dim3(C_SZ / 128, BN / 128), 256, 0, stream>>>(
      attnh, wp_t, proj_b, out, (int)BN, C_SZ, C_SZ);
}
